// MogrifierEDRNClassifier_63548336111822
// MI455X (gfx1250) — matrix operand order repaired and checked by reading, not compiled, not run on hardware
//
#include <hip/hip_runtime.h>

// ---------------------------------------------------------------------------
// MogrifierEDRNClassifier for MI455X (gfx1250, wave32, WMMA bf16 16x16x32)
//
// Shapes: B=64, T=2048, I=128, H=256, C=100, MOG_STEPS=5
//
//  1) pack_b_kernel: fp32 weight W[N,K] (used as act @ W^T) -> bf16 WMMA
//     B-fragments in the native gfx1250 lane layout.
//  2) mogrnn_recurrence: 4 persistent workgroups (16 batch rows, 8 waves).
//     State in LDS; per step 7 small GEMMs with v_wmma_f32_16x16x32_bf16;
//     N-tiles split across waves; weights live in VGPRs/L2.
//  3) mogrnn_classifier: ys @ Wfc^T + bfc, one 16x16 WMMA tile per wave.
// ---------------------------------------------------------------------------

typedef __attribute__((ext_vector_type(16))) __bf16 v16bf;
typedef __attribute__((ext_vector_type(8)))  float  v8f;

constexpr int BDIM = 64;
constexpr int TDIM = 2048;
constexpr int IDIM = 128;
constexpr int HDIM = 256;
constexpr int CDIM = 100;
constexpr int CNT  = 7;      // padded C n-tiles (112 cols)

union BF16x16 { v16bf v; uint4 q[2]; };

// B-matrix fragment (32x16 bf16), packed lane-major: lane holds 32B.
static __device__ __forceinline__ v16bf load_b_frag(const __bf16* __restrict__ frags,
                                                    int fragIdx, int lane) {
  const __bf16* p = frags + (size_t)fragIdx * 512 + lane * 16;
  BF16x16 u;
  u.q[0] = *(const uint4*)(p);
  u.q[1] = *(const uint4*)(p + 8);
  return u.v;
}

// A-matrix fragment (16x32 bf16) from a row-major [16, Kdim] bf16 buffer.
// lane L -> row M=L%16; lanes 0-15: K = kt*32+(0..7, 16..23),
//                       lanes 16-31: K = kt*32+(8..15, 24..31)
static __device__ __forceinline__ v16bf load_a_frag(const __bf16* __restrict__ base,
                                                    int Kdim, int kt, int lane) {
  const int m  = lane & 15;
  const int hi = lane >> 4;
  const __bf16* p = base + m * Kdim + kt * 32 + hi * 8;
  BF16x16 u;
  u.q[0] = *(const uint4*)(p);
  u.q[1] = *(const uint4*)(p + 16);
  return u.v;
}

static __device__ __forceinline__ v8f wmma_bf16(v16bf a, v16bf b, v8f c) {
  return __builtin_amdgcn_wmma_f32_16x16x32_bf16(false, a, false, b, (short)0, c,
                                                 false, false);
}

// Fast tanh: gfx1250 has V_TANH_F32 (transcendental unit).
static __device__ __forceinline__ float fast_tanh(float z) {
#if __has_builtin(__builtin_amdgcn_tanhf)
  return __builtin_amdgcn_tanhf(z);
#else
  // tanh(z) = 1 - 2/(exp(2z)+1); v_exp + v_rcp, no IEEE-div fixup chain
  return 1.0f - 2.0f * __builtin_amdgcn_rcpf(__expf(2.0f * z) + 1.0f);
#endif
}

// 2*sigmoid(z) == 1 + tanh(z/2)  (exact identity)
static __device__ __forceinline__ float sigmoid2(float z) {
  return 1.0f + fast_tanh(0.5f * z);
}

// ---------------------------------------------------------------------------
// Weight packer: W is [N, K] fp32 (product is act[M,K] @ W^T -> [M,N]).
// ---------------------------------------------------------------------------
__global__ void pack_b_kernel(const float* __restrict__ W, __bf16* __restrict__ dst,
                              int N, int K, int NT) {
  const int idx   = blockIdx.x * 256 + threadIdx.x;
  const int total = (K >> 5) * NT * 512;
  if (idx >= total) return;
  const int frag   = idx >> 9;
  const int within = idx & 511;
  const int lane   = within >> 4;
  const int e      = within & 15;
  const int kt = frag / NT;
  const int nt = frag - kt * NT;
  const int k = kt * 32 + ((lane >> 4) << 3) + (e & 7) + ((e >> 3) << 4);
  const int n = nt * 16 + (lane & 15);
  const float val = (n < N) ? W[n * K + k] : 0.0f;
  dst[idx] = (__bf16)val;
}

// ---------------------------------------------------------------------------
// Persistent recurrence: blockIdx.x = batch group (16 rows), 256 threads.
// ---------------------------------------------------------------------------
__global__ void __launch_bounds__(256)
mogrnn_recurrence(const float* __restrict__ x,
                  const __bf16* __restrict__ Qp,  const __bf16* __restrict__ Rp,
                  const __bf16* __restrict__ Wx0p, const __bf16* __restrict__ Wh0p,
                  const __bf16* __restrict__ Wx1p, const __bf16* __restrict__ Wh1p,
                  const float* __restrict__ b0,   const float* __restrict__ b1,
                  __bf16* __restrict__ ys) {
  __shared__ __align__(16) float  x_f[16 * IDIM];     //  8 KB
  __shared__ __align__(16) __bf16 x_b[16 * IDIM];     //  4 KB
  __shared__ __align__(16) float  h_f[16 * HDIM];     // 16 KB
  __shared__ __align__(16) __bf16 h_b[16 * HDIM];     //  8 KB
  __shared__ __align__(16) __bf16 h0_b[2][16 * HDIM]; // 16 KB

  const int tid  = threadIdx.x;
  const int lane = tid & 31;
  const int w    = tid >> 5;     // wave 0..7
  const int hi   = lane >> 4;
  const int nlo  = lane & 15;
  const int bg   = blockIdx.x;   // batch group 0..3
  const int nt0  = 2 * w;
  const int nt1  = 2 * w + 1;

  for (int i = tid; i < 16 * HDIM; i += 256) {
    h_f[i] = 0.0f;
    h_b[i] = (__bf16)0.0f;
    h0_b[0][i] = (__bf16)0.0f;
    h0_b[1][i] = (__bf16)0.0f;
  }
  __syncthreads();

  const int xrow = tid >> 4;        // 0..15
  const int xc0  = (tid & 15) * 8;  // 0..120

  for (int t = 0; t < TDIM; ++t) {
    // ---- Phase 0: stage x[bg*16 + r, t, :] into LDS (f32 + bf16) ----
    {
      const float* src = x + (((size_t)(bg * 16 + xrow) * TDIM + t) * IDIM + xc0);
#pragma unroll
      for (int j = 0; j < 8; ++j) {
        const float v = src[j];
        x_f[xrow * IDIM + xc0 + j] = v;
        x_b[xrow * IDIM + xc0 + j] = (__bf16)v;
      }
      if (t + 1 < TDIM) __builtin_prefetch(src + IDIM, 0, 1);
    }
    __syncthreads();

    // ---- Mogrifier: s=0,2,4 gate x with 2*sigmoid(h@Q^T); s=1,3 gate h ----
#pragma unroll
    for (int s = 0; s < 5; ++s) {
      if ((s & 1) == 0) {
        // G = h @ Q^T : [16,128], 8 n-tiles -> one per wave; K = H (8 k-tiles)
        v8f acc = {};
#pragma unroll
        for (int kt = 0; kt < 8; ++kt)
          acc = wmma_bf16(load_a_frag(h_b, HDIM, kt, lane),
                          load_b_frag(Qp, kt * 8 + w, lane), acc);
        const int col = w * 16 + nlo;
#pragma unroll
        for (int r = 0; r < 8; ++r) {
          const int idx = (r + 8 * hi) * IDIM + col;
          const float nx = sigmoid2(acc[r]) * x_f[idx];
          x_f[idx] = nx;
          x_b[idx] = (__bf16)nx;
        }
      } else {
        // S = x @ R^T : [16,256]; kt-outer, one A load feeds both n-tiles
        v8f acc0 = {}, acc1 = {};
#pragma unroll
        for (int kt = 0; kt < 4; ++kt) {
          const v16bf a = load_a_frag(x_b, IDIM, kt, lane);
          acc0 = wmma_bf16(a, load_b_frag(Rp, kt * 16 + nt0, lane), acc0);
          acc1 = wmma_bf16(a, load_b_frag(Rp, kt * 16 + nt1, lane), acc1);
        }
#pragma unroll
        for (int n = 0; n < 2; ++n) {
          const v8f acc = n ? acc1 : acc0;
          const int col = (nt0 + n) * 16 + nlo;
#pragma unroll
          for (int r = 0; r < 8; ++r) {
            const int idx = (r + 8 * hi) * HDIM + col;
            const float nh = sigmoid2(acc[r]) * h_f[idx];
            h_f[idx] = nh;
            h_b[idx] = (__bf16)nh;
          }
        }
      }
      __syncthreads();
    }

    const __bf16* h0prev = h0_b[t & 1];
    __bf16*       h0next = h0_b[(t & 1) ^ 1];

    // ---- h0n = tanh(xm @ Wx0^T + h0 @ Wh0^T + b0) ----
    {
      v8f acc0 = {}, acc1 = {};
#pragma unroll
      for (int kt = 0; kt < 4; ++kt) {
        const v16bf a = load_a_frag(x_b, IDIM, kt, lane);
        acc0 = wmma_bf16(a, load_b_frag(Wx0p, kt * 16 + nt0, lane), acc0);
        acc1 = wmma_bf16(a, load_b_frag(Wx0p, kt * 16 + nt1, lane), acc1);
      }
#pragma unroll
      for (int kt = 0; kt < 8; ++kt) {
        const v16bf a = load_a_frag(h0prev, HDIM, kt, lane);
        acc0 = wmma_bf16(a, load_b_frag(Wh0p, kt * 16 + nt0, lane), acc0);
        acc1 = wmma_bf16(a, load_b_frag(Wh0p, kt * 16 + nt1, lane), acc1);
      }
#pragma unroll
      for (int n = 0; n < 2; ++n) {
        const v8f acc = n ? acc1 : acc0;
        const int col = (nt0 + n) * 16 + nlo;
        const float bias = b0[col];
#pragma unroll
        for (int r = 0; r < 8; ++r)
          h0next[(r + 8 * hi) * HDIM + col] = (__bf16)fast_tanh(acc[r] + bias);
      }
    }
    __syncthreads();

    // ---- h1n = tanh(h0n @ Wx1^T + hm @ Wh1^T + b1) ----
    v8f acc0 = {}, acc1 = {};
#pragma unroll
    for (int kt = 0; kt < 8; ++kt) {
      const v16bf a = load_a_frag(h0next, HDIM, kt, lane);
      acc0 = wmma_bf16(a, load_b_frag(Wx1p, kt * 16 + nt0, lane), acc0);
      acc1 = wmma_bf16(a, load_b_frag(Wx1p, kt * 16 + nt1, lane), acc1);
    }
#pragma unroll
    for (int kt = 0; kt < 8; ++kt) {
      const v16bf a = load_a_frag(h_b, HDIM, kt, lane);
      acc0 = wmma_bf16(a, load_b_frag(Wh1p, kt * 16 + nt0, lane), acc0);
      acc1 = wmma_bf16(a, load_b_frag(Wh1p, kt * 16 + nt1, lane), acc1);
    }
    __syncthreads();  // all waves done reading h_b before we overwrite it
#pragma unroll
    for (int n = 0; n < 2; ++n) {
      const v8f acc = n ? acc1 : acc0;
      const int col = (nt0 + n) * 16 + nlo;
      const float bias = b1[col];
#pragma unroll
      for (int r = 0; r < 8; ++r) {
        const int m = r + 8 * hi;
        const float v = fast_tanh(acc[r] + bias);
        const __bf16 bv = (__bf16)v;
        h_f[m * HDIM + col] = v;
        h_b[m * HDIM + col] = bv;
        ys[((size_t)t * BDIM + bg * 16 + m) * HDIM + col] = bv;  // row = t*64+b
      }
    }
    __syncthreads();
  }
}

// ---------------------------------------------------------------------------
// Classifier: out[b,t,c] = ys[t*64+b, :] @ Wfc^T + bfc.  One wave per 16x16 tile.
// ---------------------------------------------------------------------------
__global__ void __launch_bounds__(32)
mogrnn_classifier(const __bf16* __restrict__ ys, const __bf16* __restrict__ Wfcp,
                  const float* __restrict__ bfc, float* __restrict__ out) {
  const int lane = threadIdx.x & 31;
  const int mt   = blockIdx.x;   // 0..8191
  const int nt   = blockIdx.y;   // 0..6
  const __bf16* abase = ys + (size_t)mt * 16 * HDIM;

  v8f acc = {};
#pragma unroll
  for (int kt = 0; kt < 8; ++kt)
    acc = wmma_bf16(load_a_frag(abase, HDIM, kt, lane),
                    load_b_frag(Wfcp, kt * CNT + nt, lane), acc);

  const int col = nt * 16 + (lane & 15);
  const int hi  = lane >> 4;
  if (col < CDIM) {
    const float bias = bfc[col];
#pragma unroll
    for (int r = 0; r < 8; ++r) {
      const int g  = mt * 16 + r + 8 * hi;  // linear row = t*64 + b
      const int tt = g >> 6;
      const int b  = g & 63;
      out[((size_t)b * TDIM + tt) * CDIM + col] = acc[r] + bias;
    }
  }
}

// ---------------------------------------------------------------------------
extern "C" void kernel_launch(void* const* d_in, const int* in_sizes, int n_in,
                              void* d_out, int out_size, void* d_ws, size_t ws_size,
                              hipStream_t stream) {
  (void)in_sizes; (void)n_in; (void)out_size; (void)ws_size;

  const float* x   = (const float*)d_in[0];
  const float* Q   = (const float*)d_in[1];   // [I,H]
  const float* R   = (const float*)d_in[2];   // [H,I]
  const float* Wx0 = (const float*)d_in[3];   // [H,I]
  const float* Wh0 = (const float*)d_in[4];   // [H,H]
  const float* b0  = (const float*)d_in[5];
  const float* Wx1 = (const float*)d_in[6];   // [H,H]
  const float* Wh1 = (const float*)d_in[7];   // [H,H]
  const float* b1  = (const float*)d_in[8];
  const float* Wfc = (const float*)d_in[9];   // [C,H]
  const float* bfc = (const float*)d_in[10];
  float* out = (float*)d_out;

  char* p = (char*)d_ws;
  auto carve = [&](size_t elems) {
    __bf16* r = (__bf16*)p;
    p += elems * sizeof(__bf16);
    return r;
  };
  __bf16* Qp   = carve((size_t)8 * 8  * 512);   // K=256, N=128
  __bf16* Rp   = carve((size_t)4 * 16 * 512);   // K=128, N=256
  __bf16* Wx0p = carve((size_t)4 * 16 * 512);   // K=128, N=256
  __bf16* Wh0p = carve((size_t)8 * 16 * 512);   // K=256, N=256
  __bf16* Wx1p = carve((size_t)8 * 16 * 512);
  __bf16* Wh1p = carve((size_t)8 * 16 * 512);
  __bf16* Wfcp = carve((size_t)8 * CNT * 512);  // K=256, N=100 (pad 112)
  __bf16* ys   = carve((size_t)TDIM * BDIM * HDIM);  // 64 MB

  auto pack = [&](const float* W, __bf16* dst, int N, int K, int NT) {
    const int total = (K / 32) * NT * 512;
    pack_b_kernel<<<(total + 255) / 256, 256, 0, stream>>>(W, dst, N, K, NT);
  };
  pack(Q,   Qp,   IDIM, HDIM, 8);
  pack(R,   Rp,   HDIM, IDIM, 16);
  pack(Wx0, Wx0p, HDIM, IDIM, 16);
  pack(Wh0, Wh0p, HDIM, HDIM, 16);
  pack(Wx1, Wx1p, HDIM, HDIM, 16);
  pack(Wh1, Wh1p, HDIM, HDIM, 16);
  pack(Wfc, Wfcp, CDIM, HDIM, CNT);

  mogrnn_recurrence<<<4, 256, 0, stream>>>(x, Qp, Rp, Wx0p, Wh0p, Wx1p, Wh1p,
                                           b0, b1, ys);

  const int mtiles = (TDIM * BDIM) / 16;  // 8192
  mogrnn_classifier<<<dim3(mtiles, CNT), 32, 0, stream>>>(ys, Wfcp, bfc, out);
}
